// APPNPModel_86260123173449
// MI455X (gfx1250) — matrix operand order repaired and checked by reading, not compiled, not run on hardware
//
#include <hip/hip_runtime.h>
#include <math.h>

// ---------------------------------------------------------------------------
// APPNP GNN inference for MI455X (gfx1250, wave32).
//
// Stages:
//   1. h = segment_sum(val * W1[col], row)            -- COO SpMM via atomics
//   2. z = relu(h + b1) @ W2 + b2                     -- v_wmma_f32_16x16x32_bf16
//   3. p <- 0.9 * scatter(w * p[dst], src) + 0.1 * z  -- x10, L2-resident
//   4. out = log_softmax(p)
//
// Memory-bound (~17 GB moved, ~9 GFLOP): entire working set (~140 MB) fits in
// the 192 MB global L2, so gather/scatter rounds run at L2 bandwidth.
// Workspace: h/agg (25.6 MB) + z (25.6 MB) in d_ws; d_out serves as p buffer.
// ---------------------------------------------------------------------------

#define N_NODES  100000
#define N_FEATS  2000
#define NNZ_FEAT 2000000
#define N_EDGES  3200000
#define HIDDEN   64
#define LABELS   64
#define ALPHA    0.1f
#define K_ITER   10

typedef __attribute__((ext_vector_type(16))) __bf16 v16bf;
typedef __attribute__((ext_vector_type(8)))  float  v8f;

// float -> bf16, round-to-nearest-even, via bit ops (no reliance on __bf16 arith).
__device__ __forceinline__ __bf16 f2bf(float f) {
    union { float f; unsigned u; } v; v.f = f;
    unsigned r = v.u + 0x7FFFu + ((v.u >> 16) & 1u);
    unsigned short h = (unsigned short)(r >> 16);
    return __builtin_bit_cast(__bf16, h);
}

// ---------------------------------------------------------------------------
// Stage 1: SpMM scatter. 64 threads cooperate on one nnz -> coalesced 256 B
// read of W1 row and coalesced atomic adds into h row.
// ---------------------------------------------------------------------------
__global__ void spmm_scatter(const int* __restrict__ rows,
                             const int* __restrict__ cols,
                             const float* __restrict__ vals,
                             const float* __restrict__ W1,
                             float* __restrict__ h) {
    int idx = blockIdx.x * blockDim.x + threadIdx.x;   // < 128M, fits int
    int e = idx >> 6;
    int j = idx & 63;
    if (e >= NNZ_FEAT) return;
    int r = rows[e];
    int c = cols[e];
    float v = vals[e];
    atomicAdd(&h[r * HIDDEN + j], v * W1[c * HIDDEN + j]);
}

// ---------------------------------------------------------------------------
// Stage 2: z = relu(h + b1) @ W2 + b2 via WMMA bf16 (f32 accumulate).
// One wave per 16-row tile of h. 2 K-steps (K=32 each) x 4 N-tiles of 16.
// Fragment layouts per CDNA5 ISA 7.12.2:
//   A 16x32 bf16 : lane L holds M = L&15; khalf = L>>4 selects K sub-ranges
//                  VGPR r -> K = 32s + (r>=4 ? 16 : 0) + 8*khalf + 2*(r&3)
//   B 32x16 bf16 : lane L holds N = L&15; VGPR r -> K = 32s + (r>=4 ? 16 : 0) + 8*khalf + 2*(r&3)
//   C/D 16x16 f32: lane L holds N = L&15; VGPR r -> M = r + 8*khalf
// ---------------------------------------------------------------------------
__global__ void gemm_relu_wmma(const float* __restrict__ h,
                               const float* __restrict__ b1,
                               const float* __restrict__ W2,
                               const float* __restrict__ b2,
                               float* __restrict__ z) {
    int wave = (blockIdx.x * blockDim.x + threadIdx.x) >> 5;   // tile index
    int lane = threadIdx.x & 31;
    const int n_tiles = N_NODES / 16;                          // 6250 exactly
    if (wave >= n_tiles) return;                               // uniform per wave

    const int m0    = wave * 16;
    const int lr    = lane & 15;     // A: row within tile; B/C: column within tile
    const int khalf = lane >> 4;

    // Accumulators init with b2 bias (constant per output column).
    v8f acc[4];
#pragma unroll
    for (int nt = 0; nt < 4; ++nt) {
        float bias = b2[nt * 16 + lr];
#pragma unroll
        for (int r = 0; r < 8; ++r) acc[nt][r] = bias;
    }

    const int arow = m0 + lr;
#pragma unroll
    for (int s = 0; s < 2; ++s) {
        // A fragment: relu(h + b1) fused into the load, f32 -> bf16.
        v16bf a;
#pragma unroll
        for (int r = 0; r < 8; ++r) {
            int k = s * 32 + ((r & 4) ? 16 : 0) + khalf * 8 + (r & 3) * 2;
            float x0 = h[arow * HIDDEN + k]     + b1[k];
            float x1 = h[arow * HIDDEN + k + 1] + b1[k + 1];
            x0 = x0 > 0.f ? x0 : 0.f;
            x1 = x1 > 0.f ? x1 : 0.f;
            a[2 * r]     = f2bf(x0);
            a[2 * r + 1] = f2bf(x1);
        }
#pragma unroll
        for (int nt = 0; nt < 4; ++nt) {
            // B fragment from W2 (row-major [HIDDEN][LABELS], L2-hot 16 KB).
            v16bf b;
            int bc = nt * 16 + lr;
#pragma unroll
            for (int r = 0; r < 8; ++r) {
                int k = s * 32 + ((r & 4) ? 16 : 0) + khalf * 8 + (r & 3) * 2;
                b[2 * r]     = f2bf(W2[k * LABELS + bc]);
                b[2 * r + 1] = f2bf(W2[(k + 1) * LABELS + bc]);
            }
            acc[nt] = __builtin_amdgcn_wmma_f32_16x16x32_bf16(
                false, a, false, b, (short)0, acc[nt], false, false);
        }
    }

    // Store D per C/D layout.
#pragma unroll
    for (int nt = 0; nt < 4; ++nt) {
        int colo = nt * 16 + lr;
#pragma unroll
        for (int r = 0; r < 8; ++r) {
            int rowo = m0 + r + khalf * 8;
            z[rowo * LABELS + colo] = acc[nt][r];
        }
    }
}

// ---------------------------------------------------------------------------
// Stage 3a: edge scatter. 64 threads per edge: coalesced gather of p[dst] row,
// coalesced atomic adds into agg[src] row. All buffers L2-resident.
// ---------------------------------------------------------------------------
__global__ void edge_scatter(const int* __restrict__ src,
                             const int* __restrict__ dst,
                             const float* __restrict__ w,
                             const float* __restrict__ p,
                             float* __restrict__ agg) {
    int idx = blockIdx.x * blockDim.x + threadIdx.x;   // < 204.8M, fits int
    int e = idx >> 6;
    int j = idx & 63;
    if (e >= N_EDGES) return;
    int s = src[e];
    int d = dst[e];
    float val = w[e] * p[d * LABELS + j];
    atomicAdd(&agg[s * LABELS + j], val);
}

// Stage 3b: p = (1-alpha)*agg + alpha*z
__global__ void combine(const float* __restrict__ agg,
                        const float* __restrict__ z,
                        float* __restrict__ p) {
    int i = blockIdx.x * blockDim.x + threadIdx.x;
    if (i >= N_NODES * LABELS) return;
    p[i] = (1.0f - ALPHA) * agg[i] + ALPHA * z[i];
}

// ---------------------------------------------------------------------------
// Stage 4: row-wise log_softmax, one thread per row (wave reads 32 rows =
// 8 KB contiguous, cache-resident across the three passes). In-place capable:
// p and out may alias (no __restrict__), each thread owns its row.
// ---------------------------------------------------------------------------
__global__ void logsoftmax_rows(const float* p, float* out) {
    int row = blockIdx.x * blockDim.x + threadIdx.x;
    if (row >= N_NODES) return;
    const float* x = p + (size_t)row * LABELS;
    float m = -INFINITY;
#pragma unroll 8
    for (int j = 0; j < LABELS; ++j) m = fmaxf(m, x[j]);
    float ssum = 0.f;
#pragma unroll 8
    for (int j = 0; j < LABELS; ++j) ssum += __expf(x[j] - m);
    float lse = m + __logf(ssum);
#pragma unroll 8
    for (int j = 0; j < LABELS; ++j)
        out[(size_t)row * LABELS + j] = x[j] - lse;
}

extern "C" void kernel_launch(void* const* d_in, const int* in_sizes, int n_in,
                              void* d_out, int out_size, void* d_ws, size_t ws_size,
                              hipStream_t stream) {
    const int*   feat_rows = (const int*)  d_in[0];
    const int*   feat_cols = (const int*)  d_in[1];
    const float* feat_vals = (const float*)d_in[2];
    const int*   edge_src  = (const int*)  d_in[3];
    const int*   edge_dst  = (const int*)  d_in[4];
    const float* edge_w    = (const float*)d_in[5];
    const float* W1        = (const float*)d_in[6];
    const float* b1        = (const float*)d_in[7];
    const float* W2        = (const float*)d_in[8];
    const float* b2        = (const float*)d_in[9];
    (void)in_sizes; (void)n_in; (void)out_size;

    const size_t node_elems = (size_t)N_NODES * HIDDEN;        // 6.4M (== N*LABELS)
    const size_t node_bytes = node_elems * sizeof(float);      // 25.6 MB

    float* h = (float*)d_ws;            // stage-1 output; reused as agg in stage 3
    float* z = h + node_elems;          // local predictions
    float* p = (float*)d_out;           // propagation state lives in d_out
    (void)ws_size;                      // uses 51.2 MB of workspace

    // Stage 1: h = 0; scatter features.
    hipMemsetAsync(h, 0, node_bytes, stream);
    spmm_scatter<<<(NNZ_FEAT * 64) / 256, 256, 0, stream>>>(
        feat_rows, feat_cols, feat_vals, W1, h);

    // Stage 2: z = relu(h + b1) @ W2 + b2 (WMMA).
    {
        int waves  = N_NODES / 16;                 // 6250 tiles, 1 wave each
        int blocks = (waves * 32 + 255) / 256;     // 782
        gemm_relu_wmma<<<blocks, 256, 0, stream>>>(h, b1, W2, b2, z);
    }

    // Stage 3: K_ITER propagation rounds. First round reads z directly.
    const float* pcur = z;
    for (int it = 0; it < K_ITER; ++it) {
        hipMemsetAsync(h, 0, node_bytes, stream);  // agg = 0 (reuses h buffer)
        edge_scatter<<<(N_EDGES * 64) / 256, 256, 0, stream>>>(
            edge_src, edge_dst, edge_w, pcur, h);
        combine<<<(int)((node_elems + 255) / 256), 256, 0, stream>>>(h, z, p);
        pcur = p;
    }

    // Stage 4: in-place log_softmax on d_out.
    logsoftmax_rows<<<(N_NODES + 255) / 256, 256, 0, stream>>>(p, (float*)d_out);
}
